// WaveAttTransformerClassifier_82214263980298
// MI455X (gfx1250) — matrix operand order repaired and checked by reading, not compiled, not run on hardware
//
#include <hip/hip_runtime.h>
#include <math.h>

// ---- problem constants ----
#define BB   64
#define TT   496
#define CIN  128
#define DMM  512
#define NHH  8
#define HDD  64
#define FFF  2048
#define NSS  3
#define NLL  2
#define BT_   (BB*TT)        // 31744
#define NSBT_ (NSS*BT_)      // 95232
#define GAMMA_ 0.1f
#define BIGV   1000000000.0f

// SWT filters (db4 decomposition lo; hi[i] = (-1)^(i+1) * lo[7-i])
__constant__ float DLO[8] = {
  -0.010597401784997278f, 0.032883011666982945f, 0.030841381835986965f,
  -0.18703481171888114f, -0.02798376941698385f, 0.6308807679295904f,
   0.7148465705525415f,   0.23037781330885523f };
__constant__ float DHI[8] = {
  -0.23037781330885523f,  0.7148465705525415f, -0.6308807679295904f,
  -0.02798376941698385f,  0.18703481171888114f, 0.030841381835986965f,
  -0.032883011666982945f,-0.010597401784997278f };

// ---- WMMA types / helpers ----
typedef __attribute__((ext_vector_type(16))) __bf16 v16bf;
typedef __attribute__((ext_vector_type(2)))  __bf16 v2bf;
typedef __attribute__((ext_vector_type(8)))  float  v8f;

union Frag16 { v16bf v; unsigned int u[8]; unsigned short h[16]; };

__device__ inline v8f v8f_zero() {
  v8f z;
#pragma unroll
  for (int i = 0; i < 8; i++) z[i] = 0.f;
  return z;
}

// pack two f32 -> two bf16 (round-to-nearest) in one uint.
// Preference: v_cvt_pk_bf16_f32 > (add,add,v_perm_b32) > scalar fallback.
__device__ inline unsigned int f2bf2(float lo, float hi) {
#if __has_builtin(__builtin_amdgcn_cvt_pk_bf16_f32)
  union { v2bf v; unsigned int u; } c;
  c.v = __builtin_amdgcn_cvt_pk_bf16_f32(lo, hi);
  return c.u;
#elif __has_builtin(__builtin_amdgcn_perm)
  unsigned int ul = __float_as_uint(lo) + 0x8000u;
  unsigned int uh = __float_as_uint(hi) + 0x8000u;
  return __builtin_amdgcn_perm(uh, ul, 0x07060302u);  // {uh[31:16], ul[31:16]}
#else
  unsigned int a = (__float_as_uint(lo) + 0x8000u) >> 16;
  unsigned int b = (__float_as_uint(hi) + 0x8000u) >> 16;
  return a | (b << 16);
#endif
}
__device__ inline unsigned short f2bf1(float f) {
  return (unsigned short)((__float_as_uint(f) + 0x8000u) >> 16);
}

__device__ inline v8f wmma_bf16(const Frag16& a, const Frag16& b, v8f c) {
  return __builtin_amdgcn_wmma_f32_16x16x32_bf16(
      false, a.v, false, b.v, (short)0, c, false, false);
}

// A-fragment: two 16B halves at ushort offsets kw and 16+kw, kw=(lane>=16)*8
__device__ inline void load_frag_a16(Frag16& f, const unsigned short* row, int kw) {
  uint4 q0 = *(const uint4*)(row + kw);
  uint4 q1 = *(const uint4*)(row + 16 + kw);
  f.u[0] = q0.x; f.u[1] = q0.y; f.u[2] = q0.z; f.u[3] = q0.w;
  f.u[4] = q1.x; f.u[5] = q1.y; f.u[6] = q1.z; f.u[7] = q1.w;
}
// B-fragment: two 16B halves at ushort offsets ko/2 and 16+ko/2, ko=(lane>=16)*16
__device__ inline void load_frag_b16(Frag16& f, const unsigned short* row, int ko) {
  uint4 q0 = *(const uint4*)(row + (ko >> 1));
  uint4 q1 = *(const uint4*)(row + 16 + (ko >> 1));
  f.u[0] = q0.x; f.u[1] = q0.y; f.u[2] = q0.z; f.u[3] = q0.w;
  f.u[4] = q1.x; f.u[5] = q1.y; f.u[6] = q1.z; f.u[7] = q1.w;
}

// =====================================================================
// Generic GEMM: C = act(A@B + bias) [+ addsrc], bf16 WMMA, f32 accum.
// A: [M,K] f32.  B: [K,N] f32 (transB=0, N%64==0) or [N,K] f32 (transB=1).
// Tile 256(M) x 64(N), K step 32, double-buffered LDS (1 barrier/step).
// 8 waves, each: 32 rows x 64 cols = 8 WMMAs per K step.
// =====================================================================
#define GBM 256
#define GBN 64
#define GBK 32
#define LSTR 40   // LDS row stride in ushorts (80B, 16B-aligned)

__global__ __launch_bounds__(256) void gemm_bf16_kernel(
    const float* __restrict__ A, const float* __restrict__ Bm,
    const float* __restrict__ bias, const float* __restrict__ addsrc,
    float* __restrict__ Cout, int M, int N, int K,
    long sA, long sB, long sC, int transB, int act)
{
  __shared__ __align__(16) unsigned short lsA[2][GBM][LSTR];
  __shared__ __align__(16) unsigned short lsB[2][GBN][LSTR];

  const int tid = threadIdx.x;
  const int w = tid >> 5, lane = tid & 31;
  const int m0 = blockIdx.y * GBM, n0 = blockIdx.x * GBN;
  const long bofsA = (long)blockIdx.z * sA;
  const long bofsB = (long)blockIdx.z * sB;
  const long bofsC = (long)blockIdx.z * sC;

  v8f acc[2][4];
#pragma unroll
  for (int mi = 0; mi < 2; mi++)
#pragma unroll
    for (int j = 0; j < 4; j++) acc[mi][j] = v8f_zero();

  const int mfrag = lane & 15;
  const int kw = (lane >> 4) << 3;
  const int ko = (lane >> 4) << 4;

  auto stage = [&](int kk0, int bi) {
    // A tile (256x32)
#pragma unroll
    for (int i = 0; i < 8; i++) {
      int q = tid + i * 256;              // 2048 float4 quads
      int r = q >> 3, kq = (q & 7) * 4;
      float4 val = make_float4(0.f, 0.f, 0.f, 0.f);
      int gr = m0 + r;
      if (gr < M) val = *(const float4*)(A + bofsA + (long)gr * K + kk0 + kq);
      *(unsigned int*)&lsA[bi][r][kq]     = f2bf2(val.x, val.y);
      *(unsigned int*)&lsA[bi][r][kq + 2] = f2bf2(val.z, val.w);
    }
    // B tile -> lsB[bi][n][k]
    if (!transB) {
#pragma unroll
      for (int i = 0; i < 2; i++) {
        int q = tid + i * 256;            // 512 quads (k=32, n fast)
        int kk = q >> 4, nq = (q & 15) * 4;
        float4 val = *(const float4*)(Bm + bofsB + (long)(kk0 + kk) * N + n0 + nq);
        lsB[bi][nq + 0][kk] = f2bf1(val.x);
        lsB[bi][nq + 1][kk] = f2bf1(val.y);
        lsB[bi][nq + 2][kk] = f2bf1(val.z);
        lsB[bi][nq + 3][kk] = f2bf1(val.w);
      }
    } else {
#pragma unroll
      for (int i = 0; i < 2; i++) {
        int q = tid + i * 256;            // 512 quads (n=64 rows, k fast)
        int n = q >> 3, kq = (q & 7) * 4;
        float4 val = make_float4(0.f, 0.f, 0.f, 0.f);
        int gn = n0 + n;
        if (gn < N) val = *(const float4*)(Bm + bofsB + (long)gn * K + kk0 + kq);
        *(unsigned int*)&lsB[bi][n][kq]     = f2bf2(val.x, val.y);
        *(unsigned int*)&lsB[bi][n][kq + 2] = f2bf2(val.z, val.w);
      }
    }
  };

  stage(0, 0);
  __syncthreads();

  int cur = 0;
  for (int k0 = 0; k0 < K; k0 += GBK) {
    int nxt = k0 + GBK;
    if (nxt < K) stage(nxt, cur ^ 1);         // overlap with this step's WMMAs
    if (nxt + GBK < K) {                      // prefetch two tiles ahead
      int gr = m0 + tid;
      if (gr < M) __builtin_prefetch(A + bofsA + (long)gr * K + nxt + GBK, 0, 0);
      if (tid < GBK && !transB)
        __builtin_prefetch(Bm + bofsB + (long)(nxt + GBK + tid) * N + n0, 0, 0);
    }

    Frag16 af[2];
    load_frag_a16(af[0], &lsA[cur][w * 32 + mfrag][0], kw);
    load_frag_a16(af[1], &lsA[cur][w * 32 + 16 + mfrag][0], kw);

#pragma unroll
    for (int j = 0; j < 4; j++) {
      Frag16 bf;
      load_frag_b16(bf, &lsB[cur][j * 16 + mfrag][0], ko);
      acc[0][j] = wmma_bf16(af[0], bf, acc[0][j]);
      acc[1][j] = wmma_bf16(af[1], bf, acc[1][j]);
    }
    __syncthreads();
    cur ^= 1;
  }

  // ---- epilogue ----
#pragma unroll
  for (int mi = 0; mi < 2; mi++) {
    const int rbase = m0 + w * 32 + mi * 16 + ((lane >> 4) << 3);
#pragma unroll
    for (int j = 0; j < 4; j++) {
      int col = n0 + j * 16 + mfrag;
      if (col >= N) continue;
      float bv = bias ? bias[col] : 0.f;
#pragma unroll
      for (int vr = 0; vr < 8; vr++) {
        int row = rbase + vr;
        if (row >= M) continue;
        float v = acc[mi][j][vr] + bv;
        if (act == 1) v = 0.5f * v * (1.f + erff(v * 0.70710678118654752f));
        long off = bofsC + (long)row * N + col;
        if (addsrc) v += addsrc[off];
        Cout[off] = v;
      }
    }
  }
}

// =====================================================================
// Fused attention: softmax(Q K^T / 8) V per (batch,head) = grid.y
// Q/K/V/O layout: [BH_batch, T, DMM] f32, head slice = head*HDD.
// Block 256 = 8 waves, 128 query rows; Q staged once, K/V double-buffered
// per 32-wide kv chunk; all fragments via ds_load_b128.
// =====================================================================
#define QSTR 72   // 144B rows (16B-aligned) for 64 k-dims
__global__ __launch_bounds__(256) void attn_kernel(
    const float* __restrict__ Q, const float* __restrict__ K,
    const float* __restrict__ V, float* __restrict__ O, int Tlen)
{
  __shared__ __align__(16) unsigned short lsQ[128][QSTR];
  __shared__ __align__(16) unsigned short lsK[2][32][QSTR];
  __shared__ __align__(16) unsigned short lsV[2][64][LSTR];
  __shared__ __align__(16) unsigned short lsP[8][16][LSTR];

  const int tid = threadIdx.x;
  const int w = tid >> 5, lane = tid & 31;
  const int bh = blockIdx.y;
  const int bb = bh / NHH, head = bh - bb * NHH;
  const long base = (long)bb * Tlen * DMM + head * HDD;
  const int tblock = blockIdx.x * 128;
  const int t0 = tblock + w * 16;
  const int mfrag = lane & 15;
  const int kw = (lane >> 4) << 3;
  const int ko = (lane >> 4) << 4;

  auto stage_kv = [&](int kv0, int bi) {
    // K chunk (32 kv rows x 64 k)
#pragma unroll
    for (int i = 0; i < 2; i++) {
      int q = tid + i * 256;        // 512 quads
      int kvr = q >> 4, kq = (q & 15) * 4;
      float4 val = make_float4(0.f, 0.f, 0.f, 0.f);
      int t = kv0 + kvr;
      if (t < Tlen) val = *(const float4*)(K + base + (long)t * DMM + kq);
      *(unsigned int*)&lsK[bi][kvr][kq]     = f2bf2(val.x, val.y);
      *(unsigned int*)&lsK[bi][kvr][kq + 2] = f2bf2(val.z, val.w);
    }
    // V chunk transposed: lsV[n(hd)][kv]
#pragma unroll
    for (int i = 0; i < 2; i++) {
      int q = tid + i * 256;        // 512 quads (kv row, n fast)
      int kvr = q >> 4, nq = (q & 15) * 4;
      float4 val = make_float4(0.f, 0.f, 0.f, 0.f);
      int t = kv0 + kvr;
      if (t < Tlen) val = *(const float4*)(V + base + (long)t * DMM + nq);
      lsV[bi][nq + 0][kvr] = f2bf1(val.x);
      lsV[bi][nq + 1][kvr] = f2bf1(val.y);
      lsV[bi][nq + 2][kvr] = f2bf1(val.z);
      lsV[bi][nq + 3][kvr] = f2bf1(val.w);
    }
  };

  // ---- stage Q tile (128 rows x 64 k) + first kv chunk ----
#pragma unroll
  for (int i = 0; i < 8; i++) {
    int q = tid + i * 256;          // 2048 quads
    int r = q >> 4, kq = (q & 15) * 4;
    float4 val = make_float4(0.f, 0.f, 0.f, 0.f);
    int t = tblock + r;
    if (t < Tlen) val = *(const float4*)(Q + base + (long)t * DMM + kq);
    *(unsigned int*)&lsQ[r][kq]     = f2bf2(val.x, val.y);
    *(unsigned int*)&lsQ[r][kq + 2] = f2bf2(val.z, val.w);
  }
  stage_kv(0, 0);
  __syncthreads();

  Frag16 qf[2];
  load_frag_a16(qf[0], &lsQ[w * 16 + mfrag][0], kw);
  load_frag_a16(qf[1], &lsQ[w * 16 + mfrag][32], kw);

  v8f oacc[4];
#pragma unroll
  for (int j = 0; j < 4; j++) oacc[j] = v8f_zero();
  float mrun[8], lrun[8];
#pragma unroll
  for (int vr = 0; vr < 8; vr++) { mrun[vr] = -3e38f; lrun[vr] = 0.f; }

  int cur = 0;
  for (int kv0 = 0; kv0 < Tlen; kv0 += 32) {
    if (kv0 + 32 < Tlen) stage_kv(kv0 + 32, cur ^ 1);   // overlap

    // ---- scores S (16 x 32) as two 16x16 tiles ----
    float sc[2][8];
#pragma unroll
    for (int j = 0; j < 2; j++) {
      v8f s = v8f_zero();
#pragma unroll
      for (int f = 0; f < 2; f++) {
        Frag16 kf;
        load_frag_b16(kf, &lsK[cur][j * 16 + mfrag][f * 32], ko);
        s = wmma_bf16(qf[f], kf, s);
      }
      int tk = kv0 + j * 16 + mfrag;
#pragma unroll
      for (int vr = 0; vr < 8; vr++)
        sc[j][vr] = (tk < Tlen) ? s[vr] * 0.125f : -3e38f;
    }

    // ---- online softmax (rows live in 16-lane halves) ----
    float mnew[8], alpha[8];
#pragma unroll
    for (int vr = 0; vr < 8; vr++) {
      float mx = fmaxf(sc[0][vr], sc[1][vr]);
#pragma unroll
      for (int d = 1; d < 16; d <<= 1) mx = fmaxf(mx, __shfl_xor(mx, d, 32));
      mnew[vr] = fmaxf(mrun[vr], mx);
      alpha[vr] = __expf(mrun[vr] - mnew[vr]);
    }
#pragma unroll
    for (int vr = 0; vr < 8; vr++) {
      float p0 = __expf(sc[0][vr] - mnew[vr]);
      float p1 = __expf(sc[1][vr] - mnew[vr]);
      sc[0][vr] = p0; sc[1][vr] = p1;
      float sum = p0 + p1;
#pragma unroll
      for (int d = 1; d < 16; d <<= 1) sum += __shfl_xor(sum, d, 32);
      lrun[vr] = lrun[vr] * alpha[vr] + sum;
      mrun[vr] = mnew[vr];
    }
#pragma unroll
    for (int j2 = 0; j2 < 4; j2++)
#pragma unroll
      for (int vr = 0; vr < 8; vr++) oacc[j2][vr] *= alpha[vr];

    // ---- re-layout P via wave-private LDS slab (C layout -> A frag) ----
    {
      int rofs = (lane >> 4) << 3;
#pragma unroll
      for (int j = 0; j < 2; j++)
#pragma unroll
        for (int vr = 0; vr < 8; vr++)
          lsP[w][rofs + vr][j * 16 + mfrag] = f2bf1(sc[j][vr]);
    }
    // same-wave DS ops are in-order: no block barrier needed here
    Frag16 pf;
    load_frag_a16(pf, &lsP[w][mfrag][0], kw);

    // ---- O += P @ V ----
#pragma unroll
    for (int j2 = 0; j2 < 4; j2++) {
      Frag16 vf;
      load_frag_b16(vf, &lsV[cur][j2 * 16 + mfrag][0], ko);
      oacc[j2] = wmma_bf16(pf, vf, oacc[j2]);
    }
    __syncthreads();   // staged buffer becomes current next iteration
    cur ^= 1;
  }

  // ---- write O / l ----
#pragma unroll
  for (int j2 = 0; j2 < 4; j2++)
#pragma unroll
    for (int vr = 0; vr < 8; vr++) {
      int t = t0 + ((lane >> 4) << 3) + vr;
      if (t < Tlen)
        O[base + (long)t * DMM + j2 * 16 + mfrag] = oacc[j2][vr] / lrun[vr];
    }
}

// =====================================================================
// SWT circular convolution level: det = circ(in, HI, step); a = circ(in, LO, step)
// =====================================================================
__global__ void swt_kernel(const float* __restrict__ in, float* __restrict__ det,
                           float* __restrict__ anext, int step)
{
  int idx = blockIdx.x * blockDim.x + threadIdx.x;
  if (idx >= BB * TT * CIN) return;
  int c = idx % CIN;
  int t = (idx / CIN) % TT;
  int b = idx / (CIN * TT);
  float alo = 0.f, ahi = 0.f;
#pragma unroll
  for (int f = 0; f < 8; f++) {
    int tt = t - f * step; if (tt < 0) tt += TT;
    float v = in[((long)b * TT + tt) * CIN + c];
    alo += v * DLO[f];
    ahi += v * DHI[f];
  }
  det[idx] = ahi;
  anext[idx] = alo;
}

// =====================================================================
// LayerNorm over DMM, wave per row
// =====================================================================
__global__ __launch_bounds__(256) void ln_kernel(
    const float* __restrict__ in, const float* __restrict__ g,
    const float* __restrict__ bta, float* __restrict__ out, int rows)
{
  int w = threadIdx.x >> 5, lane = threadIdx.x & 31;
  int r = blockIdx.x * 8 + w;
  if (r >= rows) return;
  const float* x = in + (long)r * DMM;
  float s = 0.f;
  for (int i = lane; i < DMM; i += 32) s += x[i];
#pragma unroll
  for (int d = 1; d < 32; d <<= 1) s += __shfl_xor(s, d, 32);
  float mu = s * (1.f / DMM);
  float v = 0.f;
  for (int i = lane; i < DMM; i += 32) { float dd = x[i] - mu; v += dd * dd; }
#pragma unroll
  for (int d = 1; d < 32; d <<= 1) v += __shfl_xor(v, d, 32);
  float rstd = rsqrtf(v * (1.f / DMM) + 1e-5f);
  for (int i = lane; i < DMM; i += 32)
    out[(long)r * DMM + i] = (x[i] - mu) * rstd * g[i] + bta[i];
}

// gfeat[b,d] = mean over s,t of src[(s*B+b), t, d]
__global__ void mean_st_kernel(const float* __restrict__ src, float* __restrict__ gfeat)
{
  int idx = blockIdx.x * blockDim.x + threadIdx.x;
  if (idx >= BB * DMM) return;
  int d = idx % DMM, b = idx / DMM;
  float s = 0.f;
  for (int sc = 0; sc < NSS; sc++) {
    const float* p = src + ((long)(sc * BB + b) * TT) * DMM + d;
    for (int t = 0; t < TT; t++) s += p[(long)t * DMM];
  }
  gfeat[idx] = s * (1.f / (NSS * TT));
}

__global__ void gate_kernel(const float* __restrict__ gfeat,
                            const float* __restrict__ Wg,
                            const float* __restrict__ bg,
                            float* __restrict__ gate)
{
  int b = threadIdx.x;
  if (b >= BB) return;
  float lg[NSS];
  for (int s = 0; s < NSS; s++) {
    float acc = bg[s];
    for (int d = 0; d < DMM; d++) acc += gfeat[(long)b * DMM + d] * Wg[d * NSS + s];
    lg[s] = acc;
  }
  float mx = fmaxf(lg[0], fmaxf(lg[1], lg[2]));
  float e0 = __expf(lg[0] - mx), e1 = __expf(lg[1] - mx), e2 = __expf(lg[2] - mx);
  float sum = e0 + e1 + e2;
  float g0 = e0 / sum, g1 = e1 / sum, g2 = e2 / sum;
  float s2 = fmaxf(g0 + g1 + g2, 1e-8f);
  gate[b * NSS + 0] = g0 / s2;
  gate[b * NSS + 1] = g1 / s2;
  gate[b * NSS + 2] = g2 / s2;
}

__global__ void z1_kernel(const float* __restrict__ src,
                          const float* __restrict__ gate,
                          float* __restrict__ z1)
{
  long idx = (long)blockIdx.x * blockDim.x + threadIdx.x;
  if (idx >= (long)BT_ * DMM) return;
  int b = (int)(idx / ((long)TT * DMM));
  long rem = idx - (long)b * TT * DMM;
  float v = 0.f;
#pragma unroll
  for (int s = 0; s < NSS; s++)
    v += src[((long)(s * BB + b) * TT) * DMM + rem] * gate[b * NSS + s];
  z1[idx] = v;
}

// row squared-norms, wave per row
__global__ __launch_bounds__(256) void rownorm_kernel(
    const float* __restrict__ in, float* __restrict__ out, int rows)
{
  int w = threadIdx.x >> 5, lane = threadIdx.x & 31;
  int r = blockIdx.x * 8 + w;
  if (r >= rows) return;
  const float* x = in + (long)r * DMM;
  float s = 0.f;
  for (int i = lane; i < DMM; i += 32) s += x[i] * x[i];
#pragma unroll
  for (int d = 1; d < 32; d <<= 1) s += __shfl_xor(s, d, 32);
  if (lane == 0) out[r] = s;
}

// soft-DTW anti-diagonal scan, one block per batch
__global__ __launch_bounds__(512) void dtw_kernel(
    const float* __restrict__ dotb, const float* __restrict__ nx,
    const float* __restrict__ nz, float* __restrict__ dcost)
{
  __shared__ float bufs[3][TT + 4];
  int tid = threadIdx.x, b = blockIdx.x;
  float* p2 = bufs[0];
  float* p1 = bufs[1];
  float* nw = bufs[2];
  if (tid <= TT) { p2[tid] = (tid == 0) ? 0.f : BIGV; p1[tid] = BIGV; }
  __syncthreads();
  for (int k = 2; k <= 2 * TT; k++) {
    float newv = BIGV;
    if (tid <= TT) {
      int i = tid, row = i - 1, col = k - 1 - i;
      bool valid = (row >= 0) && (row < TT) && (col >= 0) && (col < TT);
      float a = (i == 0) ? BIGV : p1[i - 1];
      float bm = p1[i];
      float c = (i == 0) ? BIGV : p2[i - 1];
      float m = fminf(a, fminf(bm, c));
      float sm = m - GAMMA_ * logf(__expf(-(a - m) / GAMMA_) +
                                   __expf(-(bm - m) / GAMMA_) +
                                   __expf(-(c - m) / GAMMA_));
      if (valid) {
        float dv = nx[b * TT + row] + nz[b * TT + col] -
                   2.f * dotb[((long)b * TT + row) * TT + col];
        newv = dv + sm;
      }
    }
    __syncthreads();
    if (tid <= TT) nw[tid] = newv;
    __syncthreads();
    float* tmp = p2; p2 = p1; p1 = nw; nw = tmp;
  }
  if (tid == 0) dcost[b] = p1[TT];
}

// h = x_proj * exp(-dcost[b]) + pos_enc(t,d)
__global__ void hinit_kernel(const float* __restrict__ xproj,
                             const float* __restrict__ dcost,
                             float* __restrict__ h)
{
  long idx = (long)blockIdx.x * blockDim.x + threadIdx.x;
  if (idx >= (long)BT_ * DMM) return;
  int d = (int)(idx % DMM);
  int t = (int)((idx / DMM) % TT);
  int b = (int)(idx / ((long)TT * DMM));
  int i2 = (d >> 1) << 1;
  float div = __expf(-(float)i2 * (9.210340371976184f / DMM)); // ln(10000)
  float ang = (float)t * div;
  float pe = (d & 1) ? __cosf(ang) : __sinf(ang);
  h[idx] = xproj[idx] * __expf(-dcost[b]) + pe;
}

// out[b] = xn[b, T-1, :] @ Wc + bc
__global__ void classify_kernel(const float* __restrict__ xn,
                                const float* __restrict__ Wc,
                                const float* __restrict__ bc,
                                float* __restrict__ out)
{
  int lane = threadIdx.x;
  const float* row = xn + ((long)blockIdx.x * TT + (TT - 1)) * DMM;
  float s = 0.f;
  for (int i = lane; i < DMM; i += 32) s += row[i] * Wc[i];
#pragma unroll
  for (int d = 1; d < 32; d <<= 1) s += __shfl_xor(s, d, 32);
  if (lane == 0) out[blockIdx.x] = s + bc[0];
}

// =====================================================================
// Host orchestration
// =====================================================================
static void launch_gemm(hipStream_t st, const float* A, const float* Bm,
                        const float* bias, const float* addsrc, float* Cq,
                        int M, int N, int K, int act = 0, int batch = 1,
                        long sA = 0, long sB = 0, long sC = 0, int transB = 0)
{
  dim3 g((N + GBN - 1) / GBN, (M + GBM - 1) / GBM, batch), blk(256);
  gemm_bf16_kernel<<<g, blk, 0, st>>>(A, Bm, bias, addsrc, Cq, M, N, K,
                                      sA, sB, sC, transB, act);
}

extern "C" void kernel_launch(void* const* d_in, const int* in_sizes, int n_in,
                              void* d_out, int out_size, void* d_ws, size_t ws_size,
                              hipStream_t stream)
{
  (void)in_sizes; (void)n_in; (void)out_size; (void)ws_size;
  const float* x      = (const float*)d_in[0];
  const float* W_in   = (const float*)d_in[1];
  const float* b_in   = (const float*)d_in[2];
  const float* Wq_s   = (const float*)d_in[3];
  const float* bq_s   = (const float*)d_in[4];
  const float* Wk_s   = (const float*)d_in[5];
  const float* bk_s   = (const float*)d_in[6];
  const float* Wv_s   = (const float*)d_in[7];
  const float* bv_s   = (const float*)d_in[8];
  const float* Wg     = (const float*)d_in[9];
  const float* bg     = (const float*)d_in[10];
  const float* wav_Wq = (const float*)d_in[11];
  const float* wav_bq = (const float*)d_in[12];
  const float* wav_Wk = (const float*)d_in[13];
  const float* wav_bk = (const float*)d_in[14];
  const float* wav_Wv = (const float*)d_in[15];
  const float* wav_bv = (const float*)d_in[16];
  const float* wav_Wo = (const float*)d_in[17];
  const float* wav_bo = (const float*)d_in[18];
  const float* ln1_g  = (const float*)d_in[19];
  const float* ln1_b  = (const float*)d_in[20];
  const float* eWq    = (const float*)d_in[21];
  const float* ebq    = (const float*)d_in[22];
  const float* eWk    = (const float*)d_in[23];
  const float* ebk    = (const float*)d_in[24];
  const float* eWv    = (const float*)d_in[25];
  const float* ebv    = (const float*)d_in[26];
  const float* eWo    = (const float*)d_in[27];
  const float* ebo    = (const float*)d_in[28];
  const float* ln2_g  = (const float*)d_in[29];
  const float* ln2_b  = (const float*)d_in[30];
  const float* eW1    = (const float*)d_in[31];
  const float* eb1    = (const float*)d_in[32];
  const float* eW2    = (const float*)d_in[33];
  const float* eb2    = (const float*)d_in[34];
  const float* fn_g   = (const float*)d_in[35];
  const float* fn_b   = (const float*)d_in[36];
  const float* Wc     = (const float*)d_in[37];
  const float* bc     = (const float*)d_in[38];
  float* out = (float*)d_out;

  // workspace layout
  const size_t S_BTC = (size_t)BB * TT * CIN;
  const size_t S_BTD = (size_t)BB * TT * DMM;
  const size_t S_NSD = (size_t)NSS * S_BTD;
  const size_t S_BTT = (size_t)BB * TT * TT;
  float* ws = (float*)d_ws;
  float* big0    = ws;
  float* big1    = big0 + S_NSD;
  float* big2    = big1 + S_NSD;
  float* big3    = big2 + S_NSD;
  float* details = big3 + S_NSD;       // NSS * S_BTC
  float* aw1     = details + NSS * S_BTC;
  float* aw2     = aw1 + S_BTC;
  float* xproj   = aw2 + S_BTC;
  float* z1b     = xproj + S_BTD;
  float* hbuf    = z1b + S_BTD;
  float* xnbuf   = hbuf + S_BTD;
  float* dotb    = xnbuf + S_BTD;      // S_BTT
  float* nx      = dotb + S_BTT;
  float* nz      = nx + BT_;
  float* gfeat   = nz + BT_;
  float* gate    = gfeat + (size_t)BB * DMM;
  float* dcost   = gate + BB * NSS;

  const int nBTC = BB * TT * CIN;
  const int THR = 256;

  // 1. x_proj = x @ W_in + b_in
  launch_gemm(stream, x, W_in, b_in, nullptr, xproj, BT_, DMM, CIN);

  // 2. SWT details (details[0]=level2, [1]=level1, [2]=level0)
  swt_kernel<<<(nBTC + THR - 1) / THR, THR, 0, stream>>>(x,   details + 2 * S_BTC, aw1, 1);
  swt_kernel<<<(nBTC + THR - 1) / THR, THR, 0, stream>>>(aw1, details + 1 * S_BTC, aw2, 2);
  swt_kernel<<<(nBTC + THR - 1) / THR, THR, 0, stream>>>(aw2, details + 0 * S_BTC, aw1, 4);

  // 3. per-scale Q/K/V projections -> stacked (NS*B, T, DM)
  for (int s = 0; s < NSS; s++) {
    launch_gemm(stream, details + s * S_BTC, Wq_s + (size_t)s * CIN * DMM,
                bq_s + s * DMM, nullptr, big0 + s * S_BTD, BT_, DMM, CIN);
    launch_gemm(stream, details + s * S_BTC, Wk_s + (size_t)s * CIN * DMM,
                bk_s + s * DMM, nullptr, big1 + s * S_BTD, BT_, DMM, CIN);
    launch_gemm(stream, details + s * S_BTC, Wv_s + (size_t)s * CIN * DMM,
                bv_s + s * DMM, nullptr, big2 + s * S_BTD, BT_, DMM, CIN);
  }

  // 4. wav MHA projections
  launch_gemm(stream, big0, wav_Wq, wav_bq, nullptr, big3, NSBT_, DMM, DMM); // Qp
  launch_gemm(stream, big1, wav_Wk, wav_bk, nullptr, big0, NSBT_, DMM, DMM); // Kp
  launch_gemm(stream, big2, wav_Wv, wav_bv, nullptr, big1, NSBT_, DMM, DMM); // Vp

  // 5. wav attention (1536 batch-heads)
  {
    dim3 g((TT + 127) / 128, NSS * BB * NHH);
    attn_kernel<<<g, 256, 0, stream>>>(big3, big0, big1, big2, TT);
  }

  // 6. output projection
  launch_gemm(stream, big2, wav_Wo, wav_bo, nullptr, big3, NSBT_, DMM, DMM);

  // 7. gating and z1
  mean_st_kernel<<<(BB * DMM + THR - 1) / THR, THR, 0, stream>>>(big3, gfeat);
  gate_kernel<<<1, 64, 0, stream>>>(gfeat, Wg, bg, gate);
  z1_kernel<<<(int)(((long)BT_ * DMM + THR - 1) / THR), THR, 0, stream>>>(big3, gate, z1b);

  // 8. soft-DTW
  launch_gemm(stream, xproj, z1b, nullptr, nullptr, dotb, TT, TT, DMM, 0,
              BB, (long)TT * DMM, (long)TT * DMM, (long)TT * TT, 1);
  rownorm_kernel<<<(BT_ + 7) / 8, 256, 0, stream>>>(xproj, nx, BT_);
  rownorm_kernel<<<(BT_ + 7) / 8, 256, 0, stream>>>(z1b, nz, BT_);
  dtw_kernel<<<BB, 512, 0, stream>>>(dotb, nx, nz, dcost);
  hinit_kernel<<<(int)(((long)BT_ * DMM + THR - 1) / THR), THR, 0, stream>>>(xproj, dcost, hbuf);

  // 9. transformer encoder layers
  for (int l = 0; l < NLL; l++) {
    ln_kernel<<<(BT_ + 7) / 8, 256, 0, stream>>>(hbuf, ln1_g + l * DMM, ln1_b + l * DMM, xnbuf, BT_);
    launch_gemm(stream, xnbuf, eWq + (size_t)l * DMM * DMM, ebq + l * DMM, nullptr, big0, BT_, DMM, DMM);
    launch_gemm(stream, xnbuf, eWk + (size_t)l * DMM * DMM, ebk + l * DMM, nullptr, big1, BT_, DMM, DMM);
    launch_gemm(stream, xnbuf, eWv + (size_t)l * DMM * DMM, ebv + l * DMM, nullptr, big2, BT_, DMM, DMM);
    {
      dim3 g((TT + 127) / 128, BB * NHH);
      attn_kernel<<<g, 256, 0, stream>>>(big0, big1, big2, big3, TT);
    }
    launch_gemm(stream, big3, eWo + (size_t)l * DMM * DMM, ebo + l * DMM, hbuf, hbuf, BT_, DMM, DMM);
    ln_kernel<<<(BT_ + 7) / 8, 256, 0, stream>>>(hbuf, ln2_g + l * DMM, ln2_b + l * DMM, xnbuf, BT_);
    launch_gemm(stream, xnbuf, eW1 + (size_t)l * DMM * FFF, eb1 + l * FFF, nullptr, big0, BT_, FFF, DMM, 1);
    launch_gemm(stream, big0, eW2 + (size_t)l * FFF * DMM, eb2 + l * DMM, hbuf, hbuf, BT_, DMM, FFF);
  }

  // 10. final LN + classifier
  ln_kernel<<<(BT_ + 7) / 8, 256, 0, stream>>>(hbuf, fn_g, fn_b, xnbuf, BT_);
  classify_kernel<<<BB, 32, 0, stream>>>(xnbuf, Wc, bc, out);
}
